// Text2Motion_Transformer_89008902242332
// MI455X (gfx1250) — compile-verified, matrix operand order checked by reading, not run on hardware
//
#include <hip/hip_runtime.h>
#include <hip/hip_bf16.h>
#include <math.h>

typedef __attribute__((ext_vector_type(16))) _Float16 v16h;
typedef __attribute__((ext_vector_type(8)))  float    v8f;

#define BSZ   16
#define TSEQ  512
#define CDIM  1024
#define HNUM  16
#define NDIM  64
#define LNUM  6
#define FCDIM 4096
#define CLIPD 512
#define NOUT  513            // NUM_VQ + 1
#define BT    (BSZ*TSEQ)     // 8192 tokens
#define GN_EPS (1e-5f*64.0f)

enum { ACT_NONE = 0, ACT_TANH = 1, ACT_SIG = 2, ACT_RELU2 = 3 };

// ---------------------------------------------------------------------------
// Weight swizzle: f32 row-major W[K][N] -> f16 WMMA-B fragment layout
//   out[((tn*kt + ks)*32 + lane)*16 + e] = W[k][n]
//   n = tn*16 + lane%16 ; half = lane/16
//   e<8 : k = ks*32 + half*8 + e ; e>=8 : k = ks*32 + 16 + half*8 + (e-8)
// Columns >= N are zero padded (so GEMM B loads need no guards).
// ---------------------------------------------------------------------------
__global__ void swizzle_kernel(const float* __restrict__ W,
                               _Float16* __restrict__ out, int K, int N)
{
    const int kt  = K >> 5;
    const int ntl = (N + 15) >> 4;
    const int tot = ntl * kt * 32;
    const int gid = blockIdx.x * 256 + threadIdx.x;
    if (gid >= tot) return;
    const int lane = gid & 31;
    const int ks   = (gid >> 5) % kt;
    const int n    = (gid / (kt * 32)) * 16 + (lane & 15);
    const int half = lane >> 4;
    _Float16* o = out + (size_t)gid * 16;
    #pragma unroll
    for (int e = 0; e < 16; ++e) {
        const int k = ks * 32 + ((e < 8) ? (half * 8 + e) : (16 + half * 8 + e - 8));
        const float v = (n < N) ? W[(size_t)k * N + n] : 0.0f;
        o[e] = (_Float16)v;
    }
}

// ---------------------------------------------------------------------------
// WMMA GEMM: Y[M,N] (+)= act( mixA(A[M,K]) @ W[K,N] + bias )
// One wave computes NT n-tiles (16 x NT*16) sharing one A fragment: NT
// back-to-back v_wmma_f32_16x16x32_f16 per K-step, no branches in the loop.
// B comes pre-swizzled f16 (contiguous 32B per lane per tile per K-step).
// mixA: a'(m,k) = a(m,k) + (a(m-1,k)-a(m,k))*mix[k], row -1 = 0 per batch.
// K must be a multiple of 32 (call sites: 32,64,128,512,1024,4096).
// ---------------------------------------------------------------------------
#define MIX4(a, p, m)                                 \
    a.x += (p.x * pscale - a.x) * m.x;                \
    a.y += (p.y * pscale - a.y) * m.y;                \
    a.z += (p.z * pscale - a.z) * m.z;                \
    a.w += (p.w * pscale - a.w) * m.w;

template <int ACT, bool ADD, int NT>
__global__ void __launch_bounds__(32) gemm16(
        const float* __restrict__ A, const _Float16* __restrict__ Wsw,
        float* __restrict__ Y, int M, int N, int K, int ldy, int Tdim,
        const float* __restrict__ mix, const float* __restrict__ bias,
        int tnbase)
{
    const int lane = threadIdx.x & 31;
    const int half = lane >> 4;          // 0..1
    const int lr   = lane & 15;          // 0..15
    const int kt   = K >> 5;             // K-steps
    const int tn0  = blockIdx.x * NT + tnbase;   // first n-tile of this wave
    const int tileM = blockIdx.y;
    const int mrow = tileM * 16 + lr;

    const float* Arow = A + (size_t)mrow * K;
    const float* Aprv = Arow;            // safe default
    float pscale = 0.0f;
    if (mix && (mrow % Tdim) > 0) { Aprv = Arow - K; pscale = 1.0f; }

    const size_t tstride = (size_t)kt * 512;             // elems per n-tile
    const _Float16* wp0 = Wsw + (size_t)tn0 * tstride + (size_t)lane * 16;

    v8f acc[NT];
    #pragma unroll
    for (int nt = 0; nt < NT; ++nt) acc[nt] = v8f{0.f,0.f,0.f,0.f,0.f,0.f,0.f,0.f};

    for (int ks = 0; ks < kt; ++ks) {
        const int kb0 = ks * 32 + half * 8;   // A elems e=0..7  -> K=kb0+e
        const int kb1 = kb0 + 16;             // A elems e=8..15 -> K=kb1+(e-8)
        float4 a0 = *(const float4*)(Arow + kb0);
        float4 a1 = *(const float4*)(Arow + kb0 + 4);
        float4 a2 = *(const float4*)(Arow + kb1);
        float4 a3 = *(const float4*)(Arow + kb1 + 4);
        if (mix) {
            float4 p0 = *(const float4*)(Aprv + kb0);
            float4 p1 = *(const float4*)(Aprv + kb0 + 4);
            float4 p2 = *(const float4*)(Aprv + kb1);
            float4 p3 = *(const float4*)(Aprv + kb1 + 4);
            float4 m0 = *(const float4*)(mix + kb0);
            float4 m1 = *(const float4*)(mix + kb0 + 4);
            float4 m2 = *(const float4*)(mix + kb1);
            float4 m3 = *(const float4*)(mix + kb1 + 4);
            MIX4(a0, p0, m0); MIX4(a1, p1, m1);
            MIX4(a2, p2, m2); MIX4(a3, p3, m3);
        }
        v16h af;
        af[0]  = (_Float16)a0.x; af[1]  = (_Float16)a0.y;
        af[2]  = (_Float16)a0.z; af[3]  = (_Float16)a0.w;
        af[4]  = (_Float16)a1.x; af[5]  = (_Float16)a1.y;
        af[6]  = (_Float16)a1.z; af[7]  = (_Float16)a1.w;
        af[8]  = (_Float16)a2.x; af[9]  = (_Float16)a2.y;
        af[10] = (_Float16)a2.z; af[11] = (_Float16)a2.w;
        af[12] = (_Float16)a3.x; af[13] = (_Float16)a3.y;
        af[14] = (_Float16)a3.z; af[15] = (_Float16)a3.w;

        const _Float16* wp = wp0 + (size_t)ks * 512;
        #pragma unroll
        for (int nt = 0; nt < NT; ++nt) {
            v16h bf = *(const v16h*)(wp + (size_t)nt * tstride);
            acc[nt] = __builtin_amdgcn_wmma_f32_16x16x32_f16(
                          false, af, false, bf, (short)0, acc[nt], false, false);
        }
        __builtin_prefetch(Arow + kb0 + 32, 0, 1);   // next K-block of A
    }

    // D layout: VGPR g -> row m = g + 8*half, col n = lane%16
    #pragma unroll
    for (int nt = 0; nt < NT; ++nt) {
        const int no = (tn0 + nt) * 16 + lr;
        if (no < N) {
            #pragma unroll
            for (int g = 0; g < 8; ++g) {
                const int mo = tileM * 16 + g + 8 * half;
                float val = acc[nt][g];
                if (bias) val += bias[no];
                if (ACT == ACT_TANH)  val = tanhf(val);
                if (ACT == ACT_SIG)   val = 1.0f / (1.0f + expf(-val));
                if (ACT == ACT_RELU2) { val = fmaxf(val, 0.0f); val = val * val; }
                float* yp = Y + (size_t)mo * ldy + no;
                if (ADD) *yp += val; else *yp = val;
            }
        }
    }
}

// Host-side dispatcher: full blocks of 4 n-tiles, plus a ragged tail launch.
template <int ACT, bool ADD>
static void launch_gemm(const float* A, const _Float16* Wsw, float* Y,
                        int M, int N, int K, int ldy, int Tdim,
                        const float* mix, const float* bias, hipStream_t s)
{
    const int ntl  = (N + 15) / 16;
    const int full = ntl / 4;
    const int rem  = ntl % 4;
    const dim3 b32(32);
    if (full)
        gemm16<ACT, ADD, 4><<<dim3(full, M / 16), b32, 0, s>>>(
            A, Wsw, Y, M, N, K, ldy, Tdim, mix, bias, 0);
    if (rem == 1)
        gemm16<ACT, ADD, 1><<<dim3(1, M / 16), b32, 0, s>>>(
            A, Wsw, Y, M, N, K, ldy, Tdim, mix, bias, full * 4);
    else if (rem == 2)
        gemm16<ACT, ADD, 2><<<dim3(1, M / 16), b32, 0, s>>>(
            A, Wsw, Y, M, N, K, ldy, Tdim, mix, bias, full * 4);
    else if (rem == 3)
        gemm16<ACT, ADD, 3><<<dim3(1, M / 16), b32, 0, s>>>(
            A, Wsw, Y, M, N, K, ldy, Tdim, mix, bias, full * 4);
}

// ---------------------------------------------------------------------------
// Token embedding for t >= 1: x[b, t, :] = tok_emb[idx[b, t-1], :]
// ---------------------------------------------------------------------------
__global__ void embed_kernel(const int* __restrict__ idx,
                             const float* __restrict__ tok_emb,
                             float* __restrict__ x)
{
    const int row = blockIdx.x;            // 0 .. B*(T-1)-1
    const int b = row / (TSEQ - 1);
    const int t = row % (TSEQ - 1) + 1;
    const int tok = idx[row];
    const float* e = tok_emb + (size_t)tok * CDIM;
    float* xr = x + ((size_t)b * TSEQ + t) * CDIM;
    #pragma unroll
    for (int i = 0; i < 4; ++i) {
        const int c = threadIdx.x + i * 256;
        xr[c] = e[c];
    }
}

// ---------------------------------------------------------------------------
// LayerNorm over C=1024, block=256 per row (in-place safe: X may equal Y)
// ---------------------------------------------------------------------------
__global__ void ln_kernel(const float* __restrict__ X, float* __restrict__ Y,
                          const float* __restrict__ gma,
                          const float* __restrict__ bta)
{
    __shared__ float red[256];
    const int row = blockIdx.x;
    const float* xr = X + (size_t)row * CDIM;
    float* yr = Y + (size_t)row * CDIM;
    float lx[4];
    float s = 0.f;
    #pragma unroll
    for (int i = 0; i < 4; ++i) { lx[i] = xr[threadIdx.x + i * 256]; s += lx[i]; }
    red[threadIdx.x] = s; __syncthreads();
    for (int o = 128; o > 0; o >>= 1) {
        if (threadIdx.x < o) red[threadIdx.x] += red[threadIdx.x + o];
        __syncthreads();
    }
    const float mu = red[0] * (1.0f / CDIM);
    __syncthreads();
    float s2 = 0.f;
    #pragma unroll
    for (int i = 0; i < 4; ++i) { float d = lx[i] - mu; s2 += d * d; }
    red[threadIdx.x] = s2; __syncthreads();
    for (int o = 128; o > 0; o >>= 1) {
        if (threadIdx.x < o) red[threadIdx.x] += red[threadIdx.x + o];
        __syncthreads();
    }
    const float rstd = rsqrtf(red[0] * (1.0f / CDIM) + 1e-5f);
    #pragma unroll
    for (int i = 0; i < 4; ++i) {
        const int c = threadIdx.x + i * 256;
        yr[c] = (lx[i] - mu) * rstd * gma[c] + bta[c];
    }
}

// ---------------------------------------------------------------------------
// decay = exp(-exp(-softplus(-(td + wraw)) - 0.5)) in place
// ---------------------------------------------------------------------------
__global__ void decay_kernel(float* __restrict__ wbuf, const float* __restrict__ td)
{
    const size_t i = (size_t)blockIdx.x * 256 + threadIdx.x;
    const int c = (int)(i % CDIM);
    const float z = td[c] + wbuf[i];
    const float w = -log1pf(expf(-z)) - 0.5f;
    wbuf[i] = expf(-expf(w));
}

// v0 handling: l==0 -> v0 = v ; else v = v + (v0 - v) * vm
__global__ void vblend_kernel(float* __restrict__ v, float* __restrict__ v0,
                              const float* __restrict__ vm, int l)
{
    const size_t i = (size_t)blockIdx.x * 256 + threadIdx.x;
    if (l == 0) { v0[i] = v[i]; }
    else { const float vv = v[i]; v[i] = vv + (v0[i] - vv) * vm[i]; }
}

// ---------------------------------------------------------------------------
// kk = normalize(k*misc_kkk) per (token,head);  k <- k*(1+(a-1)*misc_a)
// ---------------------------------------------------------------------------
__global__ void kkprep_kernel(float* __restrict__ k, const float* __restrict__ a,
                              float* __restrict__ kkout,
                              const float* __restrict__ misc_kkk,
                              const float* __restrict__ misc_a)
{
    __shared__ float red[NDIM];
    const int mh = blockIdx.x;
    const int m = mh / HNUM, h = mh % HNUM;
    const int i = threadIdx.x;
    const int c = h * NDIM + i;
    const size_t off = (size_t)m * CDIM + c;
    const float kv = k[off];
    const float kkv = kv * misc_kkk[c];
    red[i] = kkv * kkv; __syncthreads();
    for (int o = 32; o > 0; o >>= 1) {
        if (i < o) red[i] += red[i + o];
        __syncthreads();
    }
    const float nrm = sqrtf(red[0]);
    kkout[off] = kkv / fmaxf(nrm, 1e-12f);
    k[off] = kv * (1.0f + (a[off] - 1.0f) * misc_a[c]);
}

// ---------------------------------------------------------------------------
// RWKV-7 recurrence. One block per (b,h); 64 threads; thread i owns row i of
// the 64x64 state S in registers.  S = S*diag(w) + (S a) b^T + v k^T; y = S r
// ---------------------------------------------------------------------------
__global__ void __launch_bounds__(64) rwkv7_kernel(
        const float* __restrict__ r, const float* __restrict__ w,
        const float* __restrict__ k, const float* __restrict__ v,
        const float* __restrict__ a, const float* __restrict__ kk,
        float* __restrict__ y)
{
    __shared__ float sr[NDIM], sw[NDIM], sk[NDIM], sv[NDIM], sa[NDIM], sb[NDIM];
    const int b = blockIdx.x / HNUM;
    const int h = blockIdx.x % HNUM;
    const int i = threadIdx.x;
    float S[NDIM];
    #pragma unroll
    for (int j = 0; j < NDIM; ++j) S[j] = 0.0f;
    const size_t base = ((size_t)b * TSEQ) * CDIM + h * NDIM + i;
    for (int t = 0; t < TSEQ; ++t) {
        const size_t off = base + (size_t)t * CDIM;
        __syncthreads();
        sr[i] = r[off]; sw[i] = w[off]; sk[i] = k[off]; sv[i] = v[off];
        const float kkv = kk[off];
        sa[i] = -kkv; sb[i] = kkv * a[off];
        __syncthreads();
        float sadot = 0.0f;
        #pragma unroll
        for (int j = 0; j < NDIM; ++j) sadot += S[j] * sa[j];
        const float vi = sv[i];
        float yi = 0.0f;
        #pragma unroll
        for (int j = 0; j < NDIM; ++j) {
            const float sj = S[j] * sw[j] + sadot * sb[j] + vi * sk[j];
            S[j] = sj;
            yi += sj * sr[j];
        }
        y[off] = yi;
    }
}

// ---------------------------------------------------------------------------
// Post: yg = ( GN(y)*lnx_g + lnx_b + (sum_n r*k*faaaa) * v ) * g
// ---------------------------------------------------------------------------
__global__ void post_kernel(const float* __restrict__ y, const float* __restrict__ r,
                            const float* __restrict__ k, const float* __restrict__ v,
                            const float* __restrict__ gat,
                            const float* __restrict__ faaaa,
                            const float* __restrict__ lnx_g,
                            const float* __restrict__ lnx_b,
                            float* __restrict__ yg)
{
    __shared__ float red[NDIM], red2[NDIM];
    const int mh = blockIdx.x;
    const int m = mh / HNUM, h = mh % HNUM;
    const int i = threadIdx.x;
    const int c = h * NDIM + i;
    const size_t off = (size_t)m * CDIM + c;
    const float yv = y[off];
    const float rkv = r[off] * k[off] * faaaa[h * NDIM + i];
    red[i] = yv; red2[i] = rkv; __syncthreads();
    for (int o = 32; o > 0; o >>= 1) {
        if (i < o) { red[i] += red[i + o]; red2[i] += red2[i + o]; }
        __syncthreads();
    }
    const float mu = red[0] * (1.0f / NDIM);
    const float rk = red2[0];
    __syncthreads();
    const float d = yv - mu;
    red[i] = d * d; __syncthreads();
    for (int o = 32; o > 0; o >>= 1) {
        if (i < o) red[i] += red[i + o];
        __syncthreads();
    }
    const float rstd = rsqrtf(red[0] * (1.0f / NDIM) + GN_EPS);
    const float out = d * rstd * lnx_g[c] + lnx_b[c] + rk * v[off];
    yg[off] = out * gat[off];
}

// ---------------------------------------------------------------------------
// Host-side orchestration (graph-capture safe: launches only)
// ---------------------------------------------------------------------------
extern "C" void kernel_launch(void* const* d_in, const int* in_sizes, int n_in,
                              void* d_out, int out_size, void* d_ws, size_t ws_size,
                              hipStream_t stream)
{
    (void)in_sizes; (void)n_in; (void)out_size; (void)ws_size;
    const int*   idx      = (const int*)  d_in[0];
    const float* clip     = (const float*)d_in[1];
    const float* tok_emb  = (const float*)d_in[2];
    const float* cond_w   = (const float*)d_in[3];
    const float* cond_b   = (const float*)d_in[4];
    const float* ln0_g    = (const float*)d_in[5];
    const float* ln0_b    = (const float*)d_in[6];
    const float* ln1_g    = (const float*)d_in[7];
    const float* ln1_b    = (const float*)d_in[8];
    const float* ln2_g    = (const float*)d_in[9];
    const float* ln2_b    = (const float*)d_in[10];
    const float* lnx_g    = (const float*)d_in[11];
    const float* lnx_b    = (const float*)d_in[12];
    const float* lnf_g    = (const float*)d_in[13];
    const float* lnf_b    = (const float*)d_in[14];
    const float* maa_r    = (const float*)d_in[15];
    const float* maa_w    = (const float*)d_in[16];
    const float* maa_k    = (const float*)d_in[17];
    const float* maa_v    = (const float*)d_in[18];
    const float* maa_a    = (const float*)d_in[19];
    const float* maa_g    = (const float*)d_in[20];
    const float* td       = (const float*)d_in[21];
    const float* dw1      = (const float*)d_in[22];
    const float* dw2      = (const float*)d_in[23];
    const float* aw0      = (const float*)d_in[24];
    const float* aw1      = (const float*)d_in[25];
    const float* aw2      = (const float*)d_in[26];
    const float* mv0      = (const float*)d_in[27];
    const float* mv1      = (const float*)d_in[28];
    const float* mv2      = (const float*)d_in[29];
    const float* gw1      = (const float*)d_in[30];
    const float* gw2      = (const float*)d_in[31];
    const float* misc_kkk = (const float*)d_in[32];
    const float* misc_a   = (const float*)d_in[33];
    const float* faaaa    = (const float*)d_in[34];
    const float* Wr       = (const float*)d_in[35];
    const float* Wk       = (const float*)d_in[36];
    const float* Wv       = (const float*)d_in[37];
    const float* Wo       = (const float*)d_in[38];
    const float* ffn_maa  = (const float*)d_in[39];
    const float* Wkey     = (const float*)d_in[40];
    const float* Wval     = (const float*)d_in[41];
    const float* headW    = (const float*)d_in[42];
    float* out = (float*)d_out;

    // ---- workspace layout: f32 activations (S = BT*C floats = 32 MB each)
    char* ws = (char*)d_ws;
    const size_t S = (size_t)BT * CDIM * sizeof(float);
    float* x     = (float*)(ws + 0 * S);
    float* xn    = (float*)(ws + 1 * S);
    float* v0b   = (float*)(ws + 2 * S);
    float* yb    = (float*)(ws + 3 * S);
    float* ygb   = (float*)(ws + 4 * S);   // also vm buffer (disjoint in time)
    float* rb    = (float*)(ws + 5 * S);
    float* kb    = (float*)(ws + 6 * S);
    float* vb    = (float*)(ws + 7 * S);
    float* ab    = (float*)(ws + 8 * S);
    float* kkb   = (float*)(ws + 9 * S);
    float* dcy   = (float*)(ws + 10 * S);
    float* gb    = (float*)(ws + 11 * S);
    float* w1b   = (float*)(ws + 12 * S);                      // BT*64
    float* a1b   = (float*)(ws + 12 * S + 4ull * 1024 * 1024); // BT*32
    float* g1b   = (float*)(ws + 12 * S + 8ull * 1024 * 1024); // BT*128
    float* kffn  = rb;  // 128 MB alias of r,k,v,a (dead by FFN time)

    // ---- f16 swizzled weight region
    auto swsize = [](int K, int N) -> size_t {
        return (size_t)(((N + 15) / 16) * 16) * (size_t)K;   // elements
    };
    _Float16* hptr = (_Float16*)(ws + 12 * S + 16ull * 1024 * 1024);
    auto halloc = [&](size_t elems) { _Float16* p = hptr; hptr += elems; return p; };

    const size_t szCC  = swsize(CDIM, CDIM);
    const size_t szDw1 = swsize(CDIM, 64),   szDw2 = swsize(64, CDIM);
    const size_t szAw1 = swsize(CDIM, 32),   szAw2 = swsize(32, CDIM);
    const size_t szGw1 = swsize(CDIM, 128),  szGw2 = swsize(128, CDIM);
    const size_t szKey = swsize(CDIM, FCDIM), szVal = swsize(FCDIM, CDIM);

    _Float16* swWc   = halloc(swsize(CLIPD, CDIM));
    _Float16* swHead = halloc(swsize(CDIM, NOUT));
    _Float16* swWr   = halloc(szCC * LNUM);
    _Float16* swWk   = halloc(szCC * LNUM);
    _Float16* swWv   = halloc(szCC * LNUM);
    _Float16* swWo   = halloc(szCC * LNUM);
    _Float16* swDw1  = halloc(szDw1 * LNUM);
    _Float16* swDw2  = halloc(szDw2 * LNUM);
    _Float16* swAw1  = halloc(szAw1 * LNUM);
    _Float16* swAw2  = halloc(szAw2 * LNUM);
    _Float16* swMv1  = halloc(szAw1 * LNUM);
    _Float16* swMv2  = halloc(szAw2 * LNUM);
    _Float16* swGw1  = halloc(szGw1 * LNUM);
    _Float16* swGw2  = halloc(szGw2 * LNUM);
    _Float16* swKey  = halloc(szKey * LNUM);
    _Float16* swVal  = halloc(szVal * LNUM);

    auto swz = [&](const float* W, _Float16* dst, int K, int N) {
        const int tot = ((N + 15) / 16) * (K / 32) * 32;
        swizzle_kernel<<<(tot + 255) / 256, 256, 0, stream>>>(W, dst, K, N);
    };
    swz(cond_w, swWc, CLIPD, CDIM);
    swz(headW, swHead, CDIM, NOUT);
    for (int l = 0; l < LNUM; ++l) {
        swz(Wr  + (size_t)l*CDIM*CDIM,  swWr  + l*szCC,  CDIM, CDIM);
        swz(Wk  + (size_t)l*CDIM*CDIM,  swWk  + l*szCC,  CDIM, CDIM);
        swz(Wv  + (size_t)l*CDIM*CDIM,  swWv  + l*szCC,  CDIM, CDIM);
        swz(Wo  + (size_t)l*CDIM*CDIM,  swWo  + l*szCC,  CDIM, CDIM);
        swz(dw1 + (size_t)l*CDIM*64,    swDw1 + l*szDw1, CDIM, 64);
        swz(dw2 + (size_t)l*64*CDIM,    swDw2 + l*szDw2, 64, CDIM);
        swz(aw1 + (size_t)l*CDIM*32,    swAw1 + l*szAw1, CDIM, 32);
        swz(aw2 + (size_t)l*32*CDIM,    swAw2 + l*szAw2, 32, CDIM);
        swz(mv1 + (size_t)l*CDIM*32,    swMv1 + l*szAw1, CDIM, 32);
        swz(mv2 + (size_t)l*32*CDIM,    swMv2 + l*szAw2, 32, CDIM);
        swz(gw1 + (size_t)l*CDIM*128,   swGw1 + l*szGw1, CDIM, 128);
        swz(gw2 + (size_t)l*128*CDIM,   swGw2 + l*szGw2, 128, CDIM);
        swz(Wkey + (size_t)l*CDIM*FCDIM, swKey + l*szKey, CDIM, FCDIM);
        swz(Wval + (size_t)l*FCDIM*CDIM, swVal + l*szVal, FCDIM, CDIM);
    }

    const dim3 b256(256), b64(64);
    const int ewGrid = BT * CDIM / 256;

    // --- embedding: cond token (GEMM into x rows t=0) + token rows t>=1
    launch_gemm<ACT_NONE, false>(clip, swWc, x, BSZ, CDIM, CLIPD,
                                 TSEQ * CDIM, TSEQ, nullptr, cond_b, stream);
    embed_kernel<<<BSZ * (TSEQ - 1), b256, 0, stream>>>(idx, tok_emb, x);
    ln_kernel<<<BT, b256, 0, stream>>>(x, x, ln0_g, ln0_b);   // ln0 (layer 0 only)

    for (int l = 0; l < LNUM; ++l) {
        const size_t lc = (size_t)l * CDIM;

        // ln1
        ln_kernel<<<BT, b256, 0, stream>>>(x, xn, ln1_g + lc, ln1_b + lc);

        // r / k / v (token-shift mix fused into A load)
        launch_gemm<ACT_NONE, false>(xn, swWr + l*szCC, rb, BT, CDIM, CDIM,
                                     CDIM, TSEQ, maa_r + lc, nullptr, stream);
        launch_gemm<ACT_NONE, false>(xn, swWk + l*szCC, kb, BT, CDIM, CDIM,
                                     CDIM, TSEQ, maa_k + lc, nullptr, stream);
        launch_gemm<ACT_NONE, false>(xn, swWv + l*szCC, vb, BT, CDIM, CDIM,
                                     CDIM, TSEQ, maa_v + lc, nullptr, stream);

        // w path: tanh(xw@dw1)@dw2 -> decay = exp(-exp(-softplus(-(td+.))-0.5))
        launch_gemm<ACT_TANH, false>(xn, swDw1 + l*szDw1, w1b, BT, 64, CDIM,
                                     64, TSEQ, maa_w + lc, nullptr, stream);
        launch_gemm<ACT_NONE, false>(w1b, swDw2 + l*szDw2, dcy, BT, CDIM, 64,
                                     CDIM, TSEQ, nullptr, nullptr, stream);
        decay_kernel<<<ewGrid, b256, 0, stream>>>(dcy, td + lc);

        // a path: sigmoid(aw0 + (xa@aw1)@aw2)
        launch_gemm<ACT_NONE, false>(xn, swAw1 + l*szAw1, a1b, BT, 32, CDIM,
                                     32, TSEQ, maa_a + lc, nullptr, stream);
        launch_gemm<ACT_SIG, false>(a1b, swAw2 + l*szAw2, ab, BT, CDIM, 32,
                                    CDIM, TSEQ, nullptr, aw0 + lc, stream);

        // v0 mix
        if (l == 0) {
            vblend_kernel<<<ewGrid, b256, 0, stream>>>(vb, v0b, nullptr, 0);
        } else {
            launch_gemm<ACT_NONE, false>(xn, swMv1 + l*szAw1, a1b, BT, 32, CDIM,
                                         32, TSEQ, maa_v + lc, nullptr, stream);
            launch_gemm<ACT_SIG, false>(a1b, swMv2 + l*szAw2, ygb, BT, CDIM, 32,
                                        CDIM, TSEQ, nullptr, mv0 + lc, stream);
            vblend_kernel<<<ewGrid, b256, 0, stream>>>(vb, v0b, ygb, l);
        }

        // g path: sigmoid(xg@gw1)@gw2
        launch_gemm<ACT_SIG, false>(xn, swGw1 + l*szGw1, g1b, BT, 128, CDIM,
                                    128, TSEQ, maa_g + lc, nullptr, stream);
        launch_gemm<ACT_NONE, false>(g1b, swGw2 + l*szGw2, gb, BT, CDIM, 128,
                                     CDIM, TSEQ, nullptr, nullptr, stream);

        // kk normalize + k scale, recurrence, post (GN + rk*v + gate)
        kkprep_kernel<<<BT * HNUM, b64, 0, stream>>>(kb, ab, kkb,
                                                     misc_kkk + lc, misc_a + lc);
        rwkv7_kernel<<<BSZ * HNUM, b64, 0, stream>>>(rb, dcy, kb, vb, ab, kkb, yb);
        post_kernel<<<BT * HNUM, b64, 0, stream>>>(yb, rb, kb, vb, gb,
                                                   faaaa + (size_t)l * HNUM * NDIM,
                                                   lnx_g + lc, lnx_b + lc, ygb);

        // output projection, residual add
        launch_gemm<ACT_NONE, true>(ygb, swWo + l*szCC, x, BT, CDIM, CDIM,
                                    CDIM, TSEQ, nullptr, nullptr, stream);

        // channel mix (FFN): x += relu(mix(LN2(x))@Wkey)^2 @ Wval
        ln_kernel<<<BT, b256, 0, stream>>>(x, xn, ln2_g + lc, ln2_b + lc);
        launch_gemm<ACT_RELU2, false>(xn, swKey + l*szKey, kffn, BT, FCDIM, CDIM,
                                      FCDIM, TSEQ, ffn_maa + lc, nullptr, stream);
        launch_gemm<ACT_NONE, true>(kffn, swVal + l*szVal, x, BT, CDIM, FCDIM,
                                    CDIM, TSEQ, nullptr, nullptr, stream);
    }

    // final LN + head
    ln_kernel<<<BT, b256, 0, stream>>>(x, xn, lnf_g, lnf_b);
    launch_gemm<ACT_NONE, false>(xn, swHead, out, BT, NOUT, CDIM,
                                 NOUT, TSEQ, nullptr, nullptr, stream);
}
